// GPT_42348377539011
// MI455X (gfx1250) — matrix operand order repaired and checked by reading, not compiled, not run on hardware
//
#include <hip/hip_runtime.h>

// ---------------------------------------------------------------------------
// GPT-2 (124M) forward pass for gfx1250 (MI455X), wave32 + WMMA f16->f32.
// Round 4: BOUNDED template split -- full-tile GEMMs (all layer GEMMs) run
// guard-free straight-line staging; only lm_head (N=50257) keeps bounds.
// Block tile 128x64, wave tile 32x32 (4 WMMA accumulators), LDS double
// buffering, peeled software pipeline.
// ---------------------------------------------------------------------------

typedef __attribute__((ext_vector_type(16))) _Float16 v16h;
typedef __attribute__((ext_vector_type(8)))  _Float16 v8h;
typedef __attribute__((ext_vector_type(8)))  float    v8f;

constexpr int CFG_B = 4;
constexpr int CFG_T = 1024;
constexpr int CFG_C = 768;
constexpr int CFG_L = 12;
constexpr int CFG_H = 12;
constexpr int CFG_V = 50257;
constexpr int CFG_D = CFG_C / CFG_H;      // 64
constexpr int CFG_M = CFG_B * CFG_T;      // 4096 rows of activations
constexpr float LN_EPS = 1e-5f;

__device__ __forceinline__ float warp_red_sum(float v) {
#pragma unroll
  for (int o = 16; o > 0; o >>= 1) v += __shfl_xor(v, o);
  return v;
}
__device__ __forceinline__ float warp_red_max(float v) {
#pragma unroll
  for (int o = 16; o > 0; o >>= 1) v = fmaxf(v, __shfl_xor(v, o));
  return v;
}

// ---------------------------------------------------------------------------
// Tiled WMMA GEMM:  C[M,N] = act(A[M,K] * B[K,N] + bias [+ residual])
// A row-major fp32, B fp32 (row-major K x N, or N x K when transb=1).
// Block tile 128x64, BK=32, 256 threads = 8 waves in a 4x2 grid;
// wave tile 32x32 -> v8f acc[2][2], 4 v_wmma_f32_16x16x32_f16 per K-step.
// EPI: 0 = +bias, 1 = +bias then exact GELU, 2 = +bias then +residual.
// BOUNDED=false requires M%128==0, N%64==0 (all layer GEMMs).
// Requires K % 32 == 0 and K >= 32 (call sites: 768, 3072).
// ---------------------------------------------------------------------------
template <int EPI, bool BOUNDED>
__global__ __launch_bounds__(256) void gemm_wmma(
    const float* __restrict__ A, const float* __restrict__ Bm,
    const float* __restrict__ bias, const float* __restrict__ Res,
    float* __restrict__ Cc,
    int M, int N, int K, int lda, int ldb, int ldc, int transb) {
  constexpr int BM = 128, BN = 64, BK = 32;
  __shared__ alignas(64) _Float16 As[2][BM * BK];   // [m][k]
  __shared__ alignas(64) _Float16 Bs[2][BN * BK];   // [n][k] (transposed tile)

  const int t    = threadIdx.x;
  const int lane = t & 31;
  const int wave = t >> 5;
  const int wr   = wave >> 1;        // 0..3 (row of wave grid, 32 rows each)
  const int wc   = wave & 1;         // 0..1 (col of wave grid, 32 cols each)
  const int lm   = lane & 15;
  const int kh   = lane >> 4;        // lane half select
  const int bm   = blockIdx.y * BM;
  const int bn   = blockIdx.x * BN;

  // staging assignments: A -> 16 elems/thread (half row), B -> 8 elems/thread
  const int srA = t >> 1, scA = (t & 1) * 16;
  const int srB = t >> 2, scB = (t & 3) * 8;
  const int  gmA = bm + srA;
  const int  gnB = bn + srB;
  const bool aok = !BOUNDED || (gmA < M);
  const bool bok = !BOUNDED || (gnB < N);
  const float* Abase = A  + (size_t)gmA * lda + scA;   // + k0 each step
  const float* Bbase = Bm + (size_t)gnB * ldb + scB;   // transb fast path

  float ar[16], br[8];

  auto fetch = [&](int k0) {
    if (aok) {
#pragma unroll
      for (int q = 0; q < 4; ++q) {
        const float4 f = *(const float4*)(Abase + k0 + q * 4);
        ar[q * 4 + 0] = f.x; ar[q * 4 + 1] = f.y;
        ar[q * 4 + 2] = f.z; ar[q * 4 + 3] = f.w;
      }
    } else {
#pragma unroll
      for (int j = 0; j < 16; ++j) ar[j] = 0.0f;
    }
    if (bok) {
      if (transb) {
        const float4 g0 = *(const float4*)(Bbase + k0);
        const float4 g1 = *(const float4*)(Bbase + k0 + 4);
        br[0] = g0.x; br[1] = g0.y; br[2] = g0.z; br[3] = g0.w;
        br[4] = g1.x; br[5] = g1.y; br[6] = g1.z; br[7] = g1.w;
      } else {
#pragma unroll
        for (int j = 0; j < 8; ++j)
          br[j] = Bm[(size_t)(k0 + scB + j) * ldb + gnB];
      }
    } else {
#pragma unroll
      for (int j = 0; j < 8; ++j) br[j] = 0.0f;
    }
  };

  auto commit = [&](int buf) {
    v8h a0, a1, bh;
#pragma unroll
    for (int j = 0; j < 8; ++j) {
      a0[j] = (_Float16)ar[j];
      a1[j] = (_Float16)ar[j + 8];
      bh[j] = (_Float16)br[j];
    }
    *(v8h*)&As[buf][srA * BK + scA]     = a0;
    *(v8h*)&As[buf][srA * BK + scA + 8] = a1;
    *(v8h*)&Bs[buf][srB * BK + scB]     = bh;
  };

  v8f acc[2][2] = {};

  auto mma = [&](int cur) {
    v16h af[2], bf[2];
#pragma unroll
    for (int mi = 0; mi < 2; ++mi) {
      const int row = wr * 32 + mi * 16 + lm;
      const v8h lo = *(const v8h*)&As[cur][row * BK + kh * 16];
      const v8h hi = *(const v8h*)&As[cur][row * BK + kh * 16 + 8];
#pragma unroll
      for (int e = 0; e < 8; ++e) { af[mi][e] = lo[e]; af[mi][e + 8] = hi[e]; }
    }
#pragma unroll
    for (int ni = 0; ni < 2; ++ni) {
      const int col = wc * 32 + ni * 16 + lm;
      bf[ni] = *(const v16h*)&Bs[cur][col * BK + kh * 16];
    }
#pragma unroll
    for (int mi = 0; mi < 2; ++mi)
#pragma unroll
      for (int ni = 0; ni < 2; ++ni)
        acc[mi][ni] = __builtin_amdgcn_wmma_f32_16x16x32_f16(
            false, af[mi], false, bf[ni], (short)0, acc[mi][ni], false, false);
  };

  // peeled pipeline: prologue fill, straight-line steady state, drained tail
  fetch(0);
  commit(0);
  __syncthreads();

  const int nk = K / BK;
  for (int i = 0; i < nk - 1; ++i) {
    fetch((i + 1) * BK);        // global loads in flight over the WMMAs
    mma(i & 1);
    commit((i & 1) ^ 1);
    __syncthreads();
  }
  mma((nk - 1) & 1);

  // epilogue: C/D layout -> row m = kh*8 + v, col n = lm
#pragma unroll
  for (int mi = 0; mi < 2; ++mi) {
#pragma unroll
    for (int v = 0; v < 8; ++v) {
      const int gm = bm + wr * 32 + mi * 16 + kh * 8 + v;
#pragma unroll
      for (int ni = 0; ni < 2; ++ni) {
        const int gn = bn + wc * 32 + ni * 16 + lm;
        if (!BOUNDED || (gm < M && gn < N)) {
          float val = acc[mi][ni][v];
          if (bias) val += bias[gn];
          if (EPI == 1)
            val = 0.5f * val * (1.0f + erff(val * 0.70710678118654752f));
          if (EPI == 2) val += Res[(size_t)gm * ldc + gn];
          Cc[(size_t)gm * ldc + gn] = val;
        }
      }
    }
  }
}

// ---------------------------------------------------------------------------
// Embedding: x[row, c] = wte[idx[row], c] + wpe[row % T, c]
// ---------------------------------------------------------------------------
__global__ __launch_bounds__(256) void embed_kernel(
    const int* __restrict__ idx, const float* __restrict__ wte,
    const float* __restrict__ wpe, float* __restrict__ x) {
  const int row = blockIdx.x;
  const int tok = idx[row];
  const int pos = row % CFG_T;
  const size_t xb = (size_t)row * CFG_C;
  const size_t wb = (size_t)tok * CFG_C;
  const size_t pb = (size_t)pos * CFG_C;
  for (int c = threadIdx.x; c < CFG_C; c += 256)
    x[xb + c] = wte[wb + c] + wpe[pb + c];
}

// ---------------------------------------------------------------------------
// LayerNorm over last dim (C), one block per row.
// ---------------------------------------------------------------------------
__global__ __launch_bounds__(256) void ln_kernel(
    const float* __restrict__ x, const float* __restrict__ w,
    const float* __restrict__ b, float* __restrict__ y, int Cn) {
  const int row = blockIdx.x;
  const size_t base = (size_t)row * Cn;
  const int t = threadIdx.x;
  float s1 = 0.f, s2 = 0.f;
  for (int c = t; c < Cn; c += 256) {
    const float v = x[base + c];
    s1 += v; s2 += v * v;
  }
  __shared__ float r1[8], r2[8];
  __shared__ float sm, sr;
  s1 = warp_red_sum(s1);
  s2 = warp_red_sum(s2);
  if ((t & 31) == 0) { r1[t >> 5] = s1; r2[t >> 5] = s2; }
  __syncthreads();
  if (t == 0) {
    float a = 0.f, q = 0.f;
    for (int i = 0; i < 8; ++i) { a += r1[i]; q += r2[i]; }
    const float m = a / Cn;
    sm = m;
    sr = rsqrtf(q / Cn - m * m + LN_EPS);
  }
  __syncthreads();
  const float m = sm, rs = sr;
  for (int c = t; c < Cn; c += 256)
    y[base + c] = (x[base + c] - m) * rs * w[c] + b[c];
}

// ---------------------------------------------------------------------------
// Causal attention, one block per (b, h, q). Scores live in LDS (T floats).
// qkv row layout: [q(768) | k(768) | v(768)], head h at offset h*64.
// ---------------------------------------------------------------------------
__global__ __launch_bounds__(256) void attn_kernel(
    const float* __restrict__ qkv, float* __restrict__ ctx) {
  const int q  = blockIdx.x;
  const int hh = blockIdx.y;
  const int bb = blockIdx.z;
  const int t  = threadIdx.x;

  __shared__ float qs[CFG_D];
  __shared__ float sc[CFG_T];
  __shared__ float red[8];
  __shared__ float part[4][CFG_D];
  __shared__ float s_mx, s_inv;

  const size_t rowq = ((size_t)bb * CFG_T + q) * (3 * CFG_C);
  if (t < CFG_D) qs[t] = qkv[rowq + hh * CFG_D + t] * 0.125f;  // 1/sqrt(64)
  __syncthreads();

  // pass 1: scores for k in [0, q]
  float lmax = -3.4e38f;
  for (int k = t; k <= q; k += 256) {
    const float* kp =
        qkv + ((size_t)bb * CFG_T + k) * (3 * CFG_C) + CFG_C + hh * CFG_D;
    float d0 = 0.f;
#pragma unroll
    for (int d = 0; d < CFG_D; ++d) d0 += qs[d] * kp[d];
    sc[k] = d0;
    lmax = fmaxf(lmax, d0);
  }
  lmax = warp_red_max(lmax);
  if ((t & 31) == 0) red[t >> 5] = lmax;
  __syncthreads();
  if (t == 0) {
    float m = red[0];
    for (int i = 1; i < 8; ++i) m = fmaxf(m, red[i]);
    s_mx = m;
  }
  __syncthreads();
  const float mx = s_mx;

  // pass 2: exp + sum
  float lsum = 0.f;
  for (int k = t; k <= q; k += 256) {
    const float e = __expf(sc[k] - mx);
    sc[k] = e;
    lsum += e;
  }
  lsum = warp_red_sum(lsum);
  if ((t & 31) == 0) red[t >> 5] = lsum;
  __syncthreads();
  if (t == 0) {
    float s = 0.f;
    for (int i = 0; i < 8; ++i) s += red[i];
    s_inv = 1.0f / s;
  }
  __syncthreads();

  // pass 3: ctx[d] = sum_k att[k] * v[k,d]; 4 k-slices x 64 dims
  const int d  = t & 63;
  const int ks = t >> 6;
  float a = 0.f;
  for (int k = ks; k <= q; k += 4)
    a += sc[k] *
         qkv[((size_t)bb * CFG_T + k) * (3 * CFG_C) + 2 * CFG_C + hh * CFG_D + d];
  part[ks][d] = a;
  __syncthreads();
  if (t < CFG_D) {
    const float s = part[0][t] + part[1][t] + part[2][t] + part[3][t];
    ctx[((size_t)bb * CFG_T + q) * CFG_C + hh * CFG_D + t] = s * s_inv;
  }
}

// ---------------------------------------------------------------------------
// Per-row log-softmax NLL over V=50257 logits.
// ---------------------------------------------------------------------------
__global__ __launch_bounds__(256) void loss_row_kernel(
    const float* __restrict__ logits, const int* __restrict__ targets,
    float* __restrict__ nll, float* __restrict__ vld) {
  const int row = blockIdx.x;
  const float* lr = logits + (size_t)row * CFG_V;
  const int t = threadIdx.x;
  __shared__ float r[8];
  __shared__ float smx, ssum;

  float mx = -3.4e38f;
  for (int c = t; c < CFG_V; c += 256) mx = fmaxf(mx, lr[c]);
  mx = warp_red_max(mx);
  if ((t & 31) == 0) r[t >> 5] = mx;
  __syncthreads();
  if (t == 0) {
    float m = r[0];
    for (int i = 1; i < 8; ++i) m = fmaxf(m, r[i]);
    smx = m;
  }
  __syncthreads();
  const float m = smx;

  float s = 0.f;
  for (int c = t; c < CFG_V; c += 256) s += __expf(lr[c] - m);
  s = warp_red_sum(s);
  if ((t & 31) == 0) r[t >> 5] = s;
  __syncthreads();
  if (t == 0) {
    float a = 0.f;
    for (int i = 0; i < 8; ++i) a += r[i];
    ssum = a;
    const int tg = targets[row];
    if (tg < 0) { nll[row] = 0.0f; vld[row] = 0.0f; }
    else        { nll[row] = -(lr[tg] - m - logf(ssum)); vld[row] = 1.0f; }
  }
}

__global__ __launch_bounds__(256) void loss_reduce_kernel(
    const float* __restrict__ nll, const float* __restrict__ vld,
    float* __restrict__ out) {
  const int t = threadIdx.x;
  float s = 0.f, c = 0.f;
  for (int i = t; i < CFG_M; i += 256) { s += nll[i]; c += vld[i]; }
  __shared__ float rs[8], rc[8];
  s = warp_red_sum(s);
  c = warp_red_sum(c);
  if ((t & 31) == 0) { rs[t >> 5] = s; rc[t >> 5] = c; }
  __syncthreads();
  if (t == 0) {
    float a = 0.f, b = 0.f;
    for (int i = 0; i < 8; ++i) { a += rs[i]; b += rc[i]; }
    out[0] = a / fmaxf(b, 1.0f);
  }
}

// ---------------------------------------------------------------------------
extern "C" void kernel_launch(void* const* d_in, const int* in_sizes, int n_in,
                              void* d_out, int out_size, void* d_ws,
                              size_t ws_size, hipStream_t stream) {
  (void)in_sizes; (void)n_in; (void)out_size; (void)ws_size;

  const int*   idx     = (const int*)d_in[0];
  const int*   targets = (const int*)d_in[1];
  const float* wte     = (const float*)d_in[2];
  const float* wpe     = (const float*)d_in[3];
  const float* ln1_w   = (const float*)d_in[4];
  const float* ln1_b   = (const float*)d_in[5];
  const float* attn_w  = (const float*)d_in[6];
  const float* attn_b  = (const float*)d_in[7];
  const float* proj_w  = (const float*)d_in[8];
  const float* proj_b  = (const float*)d_in[9];
  const float* ln2_w   = (const float*)d_in[10];
  const float* ln2_b   = (const float*)d_in[11];
  const float* fc_w    = (const float*)d_in[12];
  const float* fc_b    = (const float*)d_in[13];
  const float* fcp_w   = (const float*)d_in[14];
  const float* fcp_b   = (const float*)d_in[15];
  const float* lnf_w   = (const float*)d_in[16];
  const float* lnf_b   = (const float*)d_in[17];

  float* ws  = (float*)d_ws;
  float* x   = ws;                                   // [4096, 768]
  float* h   = x   + (size_t)CFG_M * CFG_C;          // [4096, 768]
  float* qkv = h   + (size_t)CFG_M * CFG_C;          // [4096, 2304]
  float* ctx = qkv + (size_t)CFG_M * 3 * CFG_C;      // [4096, 768]
  float* fc  = ctx + (size_t)CFG_M * CFG_C;          // [4096, 3072]
  float* nll = fc  + (size_t)CFG_M * 4 * CFG_C;      // [4096]
  float* vld = nll + CFG_M;                          // [4096]
  float* logits = (float*)d_out;                     // [4096, 50257] + loss

  auto grid = [](int M, int N) { return dim3((N + 63) / 64, (M + 127) / 128); };

  embed_kernel<<<CFG_M, 256, 0, stream>>>(idx, wte, wpe, x);

  for (int l = 0; l < CFG_L; ++l) {
    // --- attention half ---
    ln_kernel<<<CFG_M, 256, 0, stream>>>(
        x, ln1_w + (size_t)l * CFG_C, ln1_b + (size_t)l * CFG_C, h, CFG_C);
    gemm_wmma<0, false><<<grid(CFG_M, 3 * CFG_C), 256, 0, stream>>>(
        h, attn_w + (size_t)l * CFG_C * 3 * CFG_C,
        attn_b + (size_t)l * 3 * CFG_C, nullptr, qkv,
        CFG_M, 3 * CFG_C, CFG_C, CFG_C, 3 * CFG_C, 3 * CFG_C, 0);
    attn_kernel<<<dim3(CFG_T, CFG_H, CFG_B), 256, 0, stream>>>(qkv, ctx);
    gemm_wmma<2, false><<<grid(CFG_M, CFG_C), 256, 0, stream>>>(
        ctx, proj_w + (size_t)l * CFG_C * CFG_C,
        proj_b + (size_t)l * CFG_C, x, x,
        CFG_M, CFG_C, CFG_C, CFG_C, CFG_C, CFG_C, 0);
    // --- MLP half ---
    ln_kernel<<<CFG_M, 256, 0, stream>>>(
        x, ln2_w + (size_t)l * CFG_C, ln2_b + (size_t)l * CFG_C, h, CFG_C);
    gemm_wmma<1, false><<<grid(CFG_M, 4 * CFG_C), 256, 0, stream>>>(
        h, fc_w + (size_t)l * CFG_C * 4 * CFG_C,
        fc_b + (size_t)l * 4 * CFG_C, nullptr, fc,
        CFG_M, 4 * CFG_C, CFG_C, CFG_C, 4 * CFG_C, 4 * CFG_C, 0);
    gemm_wmma<2, false><<<grid(CFG_M, CFG_C), 256, 0, stream>>>(
        fc, fcp_w + (size_t)l * 4 * CFG_C * CFG_C,
        fcp_b + (size_t)l * CFG_C, x, x,
        CFG_M, CFG_C, 4 * CFG_C, 4 * CFG_C, CFG_C, CFG_C, 0);
  }

  // final LN + tied lm_head (B = wte^T via transb=1), logits straight to d_out
  ln_kernel<<<CFG_M, 256, 0, stream>>>(x, lnf_w, lnf_b, h, CFG_C);
  gemm_wmma<0, true><<<grid(CFG_M, CFG_V), 256, 0, stream>>>(
      h, wte, nullptr, nullptr, logits,
      CFG_M, CFG_V, CFG_C, CFG_C, CFG_C, CFG_V, 1);

  loss_row_kernel<<<CFG_M, 256, 0, stream>>>(logits, targets, nll, vld);
  loss_reduce_kernel<<<1, 256, 0, stream>>>(nll, vld,
                                            logits + (size_t)CFG_M * CFG_V);
}
